// AudioEncoder_25838523253484
// MI455X (gfx1250) — matrix operand order repaired and checked by reading, not compiled, not run on hardware
//
#include <hip/hip_runtime.h>
#include <hip/hip_bf16.h>

// Problem dims (fixed by reference)
#define B_  4
#define S_  32768
#define C1_ 128
#define L1_ 16384
#define C2_ 256
#define L2_ 8192
#define H_  512
#define L3_ 4096
#define V_  2048
#define K_  4

typedef __attribute__((ext_vector_type(16))) __bf16 v16bf;
typedef __attribute__((ext_vector_type(8)))  __bf16 v8bf;
typedef __attribute__((ext_vector_type(8)))  float  v8f;

#define SHUF16(lo, hi) __builtin_shufflevector(lo, hi, 0,1,2,3,4,5,6,7,8,9,10,11,12,13,14,15)

// ---------------- conv1 (scalar, 0.12 GFLOP): audio -> f1T [B, L1+6, C1] bf16, ReLU, zero-pad rows elsewhere
__global__ void conv1_relu_t(const float* __restrict__ x, const float* __restrict__ w,
                             const float* __restrict__ bias, __bf16* __restrict__ f1T) {
  int idx = blockIdx.x * blockDim.x + threadIdx.x;
  if (idx >= B_ * C1_ * L1_) return;
  int c = idx % C1_, l = (idx / C1_) % L1_, b = idx / (C1_ * L1_);
  const float* xb = x + (size_t)b * S_;
  float acc = bias[c];
  int base = 2 * l - 3;
#pragma unroll
  for (int t = 0; t < 7; ++t) {
    int p = base + t;
    if (p >= 0 && p < S_) acc = fmaf(w[c * 7 + t], xb[p], acc);
  }
  f1T[((size_t)b * (L1_ + 6) + l + 3) * C1_ + c] = (__bf16)fmaxf(acc, 0.f);
}

// ---------------- zero the 3-row halo on both transposed activation buffers ----------------
__global__ void zero_pad(__bf16* __restrict__ f1T, __bf16* __restrict__ f2T) {
  int idx = blockIdx.x * blockDim.x + threadIdx.x;
  if (idx < B_ * 6 * C1_) {
    int c = idx % C1_, r = (idx / C1_) % 6, b = idx / (6 * C1_);
    int row = (r < 3) ? r : (L1_ + r);
    f1T[((size_t)b * (L1_ + 6) + row) * C1_ + c] = (__bf16)0.f;
  } else {
    int j = idx - B_ * 6 * C1_;
    if (j < B_ * 6 * C2_) {
      int c = j % C2_, r = (j / C2_) % 6, b = j / (6 * C2_);
      int row = (r < 3) ? r : (L2_ + r);
      f2T[((size_t)b * (L2_ + 6) + row) * C2_ + c] = (__bf16)0.f;
    }
  }
}

// ---------------- weight repack: w[co,ci,t] fp32 -> wp[t][co][ci] bf16 ----------------
__global__ void pack_w2(const float* __restrict__ w, __bf16* __restrict__ wp) {
  int idx = blockIdx.x * blockDim.x + threadIdx.x;
  if (idx >= C2_ * C1_ * 7) return;
  int t = idx % 7, ci = (idx / 7) % C1_, co = idx / (7 * C1_);
  wp[((size_t)t * C2_ + co) * C1_ + ci] = (__bf16)w[idx];
}
__global__ void pack_w3(const float* __restrict__ w, __bf16* __restrict__ wp) {
  int idx = blockIdx.x * blockDim.x + threadIdx.x;
  if (idx >= H_ * C2_ * 7) return;
  int t = idx % 7, ci = (idx / 7) % C2_, co = idx / (7 * C2_);
  wp[((size_t)t * H_ + co) * C2_ + ci] = (__bf16)w[idx];
}

// ---------------- conv2 as 7 shifted WMMA GEMMs: f1T -> f2T [B,L2+6,C2], ReLU ----------------
__global__ void __launch_bounds__(64) conv2_wmma(
    const __bf16* __restrict__ f1T, const __bf16* __restrict__ wp,
    const float* __restrict__ bias, __bf16* __restrict__ f2T) {
  int wave = (blockIdx.x * blockDim.x + threadIdx.x) >> 5;  // 0 .. B*(L2/16)-1
  int lane = threadIdx.x & 31;
  int lt = wave % (L2_ / 16);
  int b  = wave / (L2_ / 16);
  int half = lane >> 4, lx = lane & 15;
  int l = lt * 16 + lx;                       // A-row = output position
  const __bf16* xb = f1T + (size_t)b * (L1_ + 6) * C1_;

  for (int nt = 0; nt < C2_ / 16; ++nt) {
    v8f acc = {};
#pragma unroll
    for (int t = 0; t < 7; ++t) {
      const __bf16* arow = xb + (size_t)(2 * l + t) * C1_ + (half ? 16 : 0);
      const __bf16* brow = wp + ((size_t)t * C2_ + nt * 16 + lx) * C1_ + (half ? 16 : 0);
#pragma unroll
      for (int hs = 0; hs < C1_ / 32; ++hs) {
        v8bf alo = *(const v8bf*)(arow + hs * 32);
        v8bf ahi = *(const v8bf*)(arow + hs * 32 + 8);
        v16bf a = SHUF16(alo, ahi);
        v16bf bb = *(const v16bf*)(brow + hs * 32);
        acc = __builtin_amdgcn_wmma_f32_16x16x32_bf16(false, a, false, bb, (short)0, acc, false, false);
      }
    }
    float bv = bias[nt * 16 + lx];
#pragma unroll
    for (int r = 0; r < 8; ++r) {
      int lo = lt * 16 + half * 8 + r;
      f2T[((size_t)b * (L2_ + 6) + lo + 3) * C2_ + nt * 16 + lx] =
          (__bf16)fmaxf(acc[r] + bv, 0.f);
    }
  }
}

// ---------------- conv3 as 7 shifted WMMA GEMMs: f2T -> feats [B,L3,H] bf16, no ReLU ----------------
__global__ void __launch_bounds__(64) conv3_wmma(
    const __bf16* __restrict__ f2T, const __bf16* __restrict__ wp,
    const float* __restrict__ bias, __bf16* __restrict__ feats) {
  int wave = (blockIdx.x * blockDim.x + threadIdx.x) >> 5;  // 0 .. B*(L3/16)-1
  int lane = threadIdx.x & 31;
  int lt = wave % (L3_ / 16);
  int b  = wave / (L3_ / 16);
  int half = lane >> 4, lx = lane & 15;
  int l = lt * 16 + lx;
  const __bf16* xb = f2T + (size_t)b * (L2_ + 6) * C2_;

  for (int nt = 0; nt < H_ / 16; ++nt) {
    v8f acc = {};
#pragma unroll
    for (int t = 0; t < 7; ++t) {
      const __bf16* arow = xb + (size_t)(2 * l + t) * C2_ + (half ? 16 : 0);
      const __bf16* brow = wp + ((size_t)t * H_ + nt * 16 + lx) * C2_ + (half ? 16 : 0);
#pragma unroll
      for (int hs = 0; hs < C2_ / 32; ++hs) {
        v8bf alo = *(const v8bf*)(arow + hs * 32);
        v8bf ahi = *(const v8bf*)(arow + hs * 32 + 8);
        v16bf a = SHUF16(alo, ahi);
        v16bf bb = *(const v16bf*)(brow + hs * 32);
        acc = __builtin_amdgcn_wmma_f32_16x16x32_bf16(false, a, false, bb, (short)0, acc, false, false);
      }
    }
    float bv = bias[nt * 16 + lx];
#pragma unroll
    for (int r = 0; r < 8; ++r) {
      int lo = lt * 16 + half * 8 + r;
      feats[((size_t)b * L3_ + lo) * H_ + nt * 16 + lx] = (__bf16)(acc[r] + bv);
    }
  }
}

// ---------------- codebook fp32 -> bf16 ----------------
__global__ void cb_to_bf16(const float* __restrict__ cb, __bf16* __restrict__ cbb) {
  int idx = blockIdx.x * blockDim.x + threadIdx.x;
  if (idx >= K_ * V_ * H_) return;
  cbb[idx] = (__bf16)cb[idx];
}

// ---------------- c2[k,v] = sum_h cb^2 (fp32) ----------------
__global__ void c2_kernel(const float* __restrict__ cb, float* __restrict__ c2) {
  int idx = blockIdx.x * blockDim.x + threadIdx.x;
  if (idx >= K_ * V_) return;
  const float4* r = (const float4*)(cb + (size_t)idx * H_);
  float s = 0.f;
  for (int i = 0; i < H_ / 4; ++i) {
    float4 e = r[i];
    s = fmaf(e.x, e.x, s); s = fmaf(e.y, e.y, s);
    s = fmaf(e.z, e.z, s); s = fmaf(e.w, e.w, s);
  }
  c2[idx] = s;
}

// ---------------- WMMA distance + argmin, async-LDS double-buffered B staging ----------------
// Block = 8 waves, each wave owns one 16-row L tile of the same (b,k); the block shares one
// 16-column codebook tile per step, staged to LDS with global_load_async_to_lds_b128 so the
// copy of tile vt+1 overlaps the 16 WMMAs of tile vt (ASYNCcnt pipelining).
#define SROW 520                      // 512 + 8 bf16 pad: kills 16-way LDS bank conflicts
__device__ __forceinline__ void stage_tile(const __bf16* __restrict__ cbk, int vt,
                                           __bf16* sbase, int tid) {
#pragma unroll
  for (int i = 0; i < 4; ++i) {
    int o = tid + i * 256;                       // 1024 x 16B = one 16x512 bf16 tile
    int row = o >> 6;
    int c = o & 63;
    const __bf16* g = cbk + ((size_t)(vt * 16 + row)) * H_ + c * 8;
    unsigned lds = (unsigned)(uintptr_t)(sbase + row * SROW + c * 8);
    asm volatile("global_load_async_to_lds_b128 %0, %1, off"
                 :: "v"(lds), "v"(g) : "memory");
  }
}

__global__ void __launch_bounds__(256) dist_argmin(
    const __bf16* __restrict__ feats, const __bf16* __restrict__ cbb,
    const float* __restrict__ c2, int* __restrict__ tok, float* __restrict__ tokf) {
  __shared__ __bf16 sbuf[2][16 * SROW];
  int tid  = threadIdx.x;
  int wv   = tid >> 5;
  int lane = tid & 31;
  int grp   = blockIdx.x;                 // 0 .. B*K*32-1
  int tile8 = grp % 32;
  int k     = (grp / 32) % K_;
  int b     = grp / (32 * K_);
  int lt    = tile8 * 8 + wv;
  int half = lane >> 4, lx = lane & 15;

  // Preload all 16 A fragments (rows lt*16+lx, K=512); ISA 16-bit A 16x32 layout.
  const __bf16* arow = feats + ((size_t)b * L3_ + lt * 16 + lx) * H_ + (half ? 8 : 0);
  v16bf afrag[16];
#pragma unroll
  for (int hs = 0; hs < 16; ++hs) {
    v8bf alo = *(const v8bf*)(arow + hs * 32);
    v8bf ahi = *(const v8bf*)(arow + hs * 32 + 16);
    afrag[hs] = SHUF16(alo, ahi);
  }

  float bestd[8];
  int   bestv[8];
#pragma unroll
  for (int r = 0; r < 8; ++r) { bestd[r] = 3.4e38f; bestv[r] = 0; }

  const __bf16* cbk = cbb + (size_t)k * V_ * H_;
  const float*  c2k = c2 + k * V_;

  stage_tile(cbk, 0, &sbuf[0][0], tid);
  for (int vt = 0; vt < V_ / 16; ++vt) {
    asm volatile("s_wait_asynccnt 0x0" ::: "memory");  // own stage(vt) done
    __syncthreads();                                   // everyone's stage(vt) done, prior reads done
    if (vt + 1 < V_ / 16) stage_tile(cbk, vt + 1, &sbuf[(vt + 1) & 1][0], tid);

    const __bf16* bb = &sbuf[vt & 1][lx * SROW + (half ? 8 : 0)];
    v8f acc = {};
#pragma unroll
    for (int hs = 0; hs < 16; ++hs) {
      v8bf blo = *(const v8bf*)(bb + hs * 32);
      v8bf bhi = *(const v8bf*)(bb + hs * 32 + 16);
      v16bf bf = SHUF16(blo, bhi);
      acc = __builtin_amdgcn_wmma_f32_16x16x32_bf16(
          false, afrag[hs], false, bf, (short)0, acc, false, false);
    }
    int v = vt * 16 + lx;
    float c2v = c2k[v];
#pragma unroll
    for (int r = 0; r < 8; ++r) {
      float d = fmaf(-2.f, acc[r], c2v);               // c2 - 2*cross (||f||^2 const over v)
      if (d < bestd[r]) { bestd[r] = d; bestv[r] = v; }
    }
    __syncthreads();                                   // reads of buf[vt&1] done before re-stage
  }

  // Min-reduce across the 16 column-lanes of each half; first-index tie-break (jnp.argmin).
#pragma unroll
  for (int off = 8; off >= 1; off >>= 1) {
#pragma unroll
    for (int r = 0; r < 8; ++r) {
      float od = __shfl_xor(bestd[r], off, 32);
      int   ov = __shfl_xor(bestv[r], off, 32);
      if (od < bestd[r] || (od == bestd[r] && ov < bestv[r])) { bestd[r] = od; bestv[r] = ov; }
    }
  }
  if (lx == 0) {
    size_t obase = ((size_t)(b * K_ + k)) * L3_ + lt * 16 + half * 8;
#pragma unroll
    for (int r = 0; r < 8; ++r) {
      tok[obase + r]  = bestv[r];
      tokf[obase + r] = (float)bestv[r];
    }
  }
}

// ---------------- emb = mean_k embedding[tok[b,k,l]] -> [B,L3,H] ----------------
__global__ void emb_mean(const int* __restrict__ tok, const float* __restrict__ emb,
                         float* __restrict__ out) {
  int idx = blockIdx.x * blockDim.x + threadIdx.x;
  if (idx >= B_ * L3_ * (H_ / 4)) return;
  int hv = idx % (H_ / 4);
  int l  = (idx / (H_ / 4)) % L3_;
  int b  = idx / ((H_ / 4) * L3_);
  size_t tbase = (size_t)b * K_ * L3_ + l;
  float sx = 0.f, sy = 0.f, sz = 0.f, sw = 0.f;
#pragma unroll
  for (int k = 0; k < K_; ++k) {
    int t = tok[tbase + (size_t)k * L3_];
    const float4 e = *(const float4*)(emb + (size_t)t * H_ + hv * 4);
    sx += e.x; sy += e.y; sz += e.z; sw += e.w;
  }
  float4 o; o.x = sx * 0.25f; o.y = sy * 0.25f; o.z = sz * 0.25f; o.w = sw * 0.25f;
  *(float4*)(out + ((size_t)b * L3_ + l) * H_ + hv * 4) = o;
}

extern "C" void kernel_launch(void* const* d_in, const int* in_sizes, int n_in,
                              void* d_out, int out_size, void* d_ws, size_t ws_size,
                              hipStream_t stream) {
  const float* audio     = (const float*)d_in[0];
  const float* w1        = (const float*)d_in[1];
  const float* b1        = (const float*)d_in[2];
  const float* w2        = (const float*)d_in[3];
  const float* b2        = (const float*)d_in[4];
  const float* w3        = (const float*)d_in[5];
  const float* b3        = (const float*)d_in[6];
  const float* codebook  = (const float*)d_in[7];
  const float* embedding = (const float*)d_in[8];

  // Workspace layout (~58.5 MB, no aliasing):
  char* ws = (char*)d_ws;
  __bf16* f1T   = (__bf16*)(ws);                          // 4*16390*128*2 = 16,783,360
  __bf16* f2T   = (__bf16*)(ws + (size_t)16783360);       // 4*8198*256*2  = 16,789,504
  __bf16* feats = (__bf16*)(ws + (size_t)33572864);       // 16,777,216
  __bf16* cbb   = (__bf16*)(ws + (size_t)50350080);       //  8,388,608
  __bf16* wp2   = (__bf16*)(ws + (size_t)58738688);       //    458,752
  __bf16* wp3   = (__bf16*)(ws + (size_t)59197440);       //  1,835,008
  float*  c2buf = (float*) (ws + (size_t)61032448);       //     32,768
  int*    tokbuf= (int*)   (ws + (size_t)61065216);       //    262,144

  float* out    = (float*)d_out;
  float* tokf   = out;                                    // B*K*L3 tokens (as float)
  float* embout = out + (size_t)B_ * K_ * L3_;

  pack_w2<<<(C2_ * C1_ * 7 + 255) / 256, 256, 0, stream>>>(w2, wp2);
  pack_w3<<<(H_ * C2_ * 7 + 255) / 256, 256, 0, stream>>>(w3, wp3);
  cb_to_bf16<<<(K_ * V_ * H_) / 256, 256, 0, stream>>>(codebook, cbb);
  c2_kernel<<<(K_ * V_) / 256, 256, 0, stream>>>(codebook, c2buf);
  zero_pad<<<(B_ * 6 * (C1_ + C2_) + 255) / 256, 256, 0, stream>>>(f1T, f2T);

  conv1_relu_t<<<(B_ * C1_ * L1_) / 256, 256, 0, stream>>>(audio, w1, b1, f1T);
  conv2_wmma<<<(B_ * (L2_ / 16)) / 2, 64, 0, stream>>>(f1T, wp2, b2, f2T);
  conv3_wmma<<<(B_ * (L3_ / 16)) / 2, 64, 0, stream>>>(f2T, wp3, b3, feats);

  dist_argmin<<<B_ * K_ * 32, 256, 0, stream>>>(feats, cbb, c2buf, tokbuf, tokf);

  emb_mean<<<(B_ * L3_ * (H_ / 4)) / 256, 256, 0, stream>>>(tokbuf, embedding, embout);
}
